// MultiheadAttention_86535001080016
// MI455X (gfx1250) — compile-verified, matrix operand order checked by reading, not run on hardware
//
#include <hip/hip_runtime.h>

// ---------------------------------------------------------------------------
// MultiheadAttention forward for MI455X (gfx1250), wave32, WMMA bf16.
// B=2, S=2048, D=1024, H=16, DK=64.  scale = 1/sqrt(64) = 0.125
// Attention kernel stages K/V chunks into LDS with gfx1250 async copies
// (global_load_async_to_lds_b128 / ASYNCcnt), double-buffered.
// ---------------------------------------------------------------------------

typedef __attribute__((ext_vector_type(16))) __bf16 v16bf;
typedef __attribute__((ext_vector_type(8)))  __bf16 v8bf;
typedef __attribute__((ext_vector_type(8)))  float  v8f;

#define BB   2
#define SS   2048
#define DD   1024
#define HH   16
#define DKK  64
#define SCL  0.125f

// ---- WMMA wrapper: D = A(16x32 bf16) * B(32x16 bf16) + C(16x16 f32) --------
__device__ __forceinline__ v8f wmma_bf16(v16bf a, v16bf b, v8f c) {
  return __builtin_amdgcn_wmma_f32_16x16x32_bf16(
      /*neg_a=*/false, a, /*neg_b=*/false, b,
      /*c_mod=*/(short)0, c, /*reuse_a=*/false, /*reuse_b=*/false);
}

// ---- A/B fragment loaders ---------------------------------------------------
// 16-bit 16x32 A layout (ISA 7.12.2): lane = row (lane&15); lane half selects
// K sub-block:  elems 0..7 -> K = kc + khalf*8 + e
//               elems 8..15 -> K = kc + 16 + khalf*8 + (e-8)
// B (32x16) is symmetric with lane = column.
__device__ __forceinline__ v16bf frag_from_f32(const float* __restrict__ row,
                                               int kc, int khalf) {
  const float4* p0 = reinterpret_cast<const float4*>(row + kc + khalf * 8);
  const float4* p1 = reinterpret_cast<const float4*>(row + kc + 16 + khalf * 8);
  float4 x0 = p0[0], x1 = p0[1];
  float4 y0 = p1[0], y1 = p1[1];
  v16bf r;
  r[0]  = (__bf16)x0.x; r[1]  = (__bf16)x0.y; r[2]  = (__bf16)x0.z; r[3]  = (__bf16)x0.w;
  r[4]  = (__bf16)x1.x; r[5]  = (__bf16)x1.y; r[6]  = (__bf16)x1.z; r[7]  = (__bf16)x1.w;
  r[8]  = (__bf16)y0.x; r[9]  = (__bf16)y0.y; r[10] = (__bf16)y0.z; r[11] = (__bf16)y0.w;
  r[12] = (__bf16)y1.x; r[13] = (__bf16)y1.y; r[14] = (__bf16)y1.z; r[15] = (__bf16)y1.w;
  return r;
}

__device__ __forceinline__ v16bf frag_from_bf16(const __bf16* __restrict__ row,
                                                int kc, int khalf) {
  v8bf a = *reinterpret_cast<const v8bf*>(row + kc + khalf * 8);
  v8bf b = *reinterpret_cast<const v8bf*>(row + kc + 16 + khalf * 8);
  v16bf r;
#pragma unroll
  for (int i = 0; i < 8; ++i) { r[i] = a[i]; r[8 + i] = b[i]; }
  return r;
}

// ---- 16-lane-half reductions (wave32) --------------------------------------
__device__ __forceinline__ float half16_max(float v) {
#pragma unroll
  for (int off = 1; off < 16; off <<= 1) v = fmaxf(v, __shfl_xor(v, off, 32));
  return v;
}
__device__ __forceinline__ float half16_sum(float v) {
#pragma unroll
  for (int off = 1; off < 16; off <<= 1) v += __shfl_xor(v, off, 32);
  return v;
}

// ---- gfx1250 async copy: global -> LDS, 16B per lane, tracked by ASYNCcnt --
__device__ __forceinline__ void async_ld_b128(unsigned int lds_off,
                                              const void* gaddr) {
  asm volatile("global_load_async_to_lds_b128 %0, %1, off"
               :: "v"(lds_off),
                  "v"((unsigned long long)(uintptr_t)gaddr)
               : "memory");
}
__device__ __forceinline__ void wait_async_le(int n) {
  if (n == 0) asm volatile("s_wait_asynccnt 0" ::: "memory");
  else        asm volatile("s_wait_asynccnt 4" ::: "memory");
}

// ===========================================================================
// Kernel 1: y = x @ W^T + b   (fp32 in, bf16 out).
// One wave -> 16x32 output tile.  vmode 0: dst[B,H,S,DK]; vmode 1: dst[B,H,DK,S]
// Tiles: 256 mTiles (tokens) x 32 nPairs -> 8192 tiles, 4 waves/block.
// ===========================================================================
__global__ __launch_bounds__(128) void proj_qkv(
    const float* __restrict__ x, const float* __restrict__ w,
    const float* __restrict__ bias, unsigned short* __restrict__ dst_u,
    int vmode) {
  __bf16* dst = reinterpret_cast<__bf16*>(dst_u);
  const int lane = threadIdx.x & 31;
  const int wv   = threadIdx.x >> 5;
  const int tile = blockIdx.x * 4 + wv;      // 0..8191
  const int mT   = tile >> 5;                // 0..255
  const int nP   = tile & 31;                // 0..31 (pair of 16-col tiles)
  const int col  = lane & 15, hi = lane >> 4;

  const float* xrow  = x + (size_t)(mT * 16 + col) * DD;
  const float* wrow0 = w + (size_t)(nP * 32 + col) * DD;
  const float* wrow1 = wrow0 + (size_t)16 * DD;

  v8f c0 = {}, c1 = {};
  for (int kc = 0; kc < DD; kc += 32) {
    v16bf a  = frag_from_f32(xrow,  kc, hi);
    v16bf b0 = frag_from_f32(wrow0, kc, hi);
    v16bf b1 = frag_from_f32(wrow1, kc, hi);
    c0 = wmma_bf16(a, b0, c0);
    c1 = wmma_bf16(a, b1, c1);
  }
  const float bv0 = bias[nP * 32 + col];
  const float bv1 = bias[nP * 32 + 16 + col];
  const int f0 = nP * 32 + col, f1 = f0 + 16;
#pragma unroll
  for (int r = 0; r < 8; ++r) {
    const int token = mT * 16 + r + 8 * hi;      // C layout: row = r + 8*half
    const int b = token >> 11, s = token & (SS - 1);
    const float v0 = c0[r] + bv0, v1 = c1[r] + bv1;
    if (vmode == 0) {
      dst[(((size_t)b * HH + (f0 >> 6)) * SS + s) * DKK + (f0 & 63)] = (__bf16)v0;
      dst[(((size_t)b * HH + (f1 >> 6)) * SS + s) * DKK + (f1 & 63)] = (__bf16)v1;
    } else {  // V stored transposed: [B,H,DK,S]
      dst[(((size_t)b * HH + (f0 >> 6)) * DKK + (f0 & 63)) * SS + s] = (__bf16)v0;
      dst[(((size_t)b * HH + (f1 >> 6)) * DKK + (f1 & 63)) * SS + s] = (__bf16)v1;
    }
  }
}

// ===========================================================================
// Kernel 2: flash attention.  4 waves/block = 4 consecutive 16-row q tiles of
// the SAME (b,h): the block cooperatively async-copies each 32-key K/V chunk
// into LDS (double-buffered), then every wave feeds WMMA from LDS.
//   LDS: kbuf 2x[32 kv][64 dk] (4KB each, contiguous copy of global chunk)
//        vbuf 2x[64 dk][32 kv] (4KB each, 64B rows of V^T)
//        ptile 4x[16][32]      (per-wave P staging, 1KB each)
// ===========================================================================
__global__ __launch_bounds__(128) void attn_fwd(
    const unsigned short* __restrict__ q_u, const unsigned short* __restrict__ k_u,
    const unsigned short* __restrict__ vt_u,
    const unsigned char* __restrict__ pad, const unsigned char* __restrict__ am,
    unsigned short* __restrict__ ao_u) {
  const __bf16* Qh = reinterpret_cast<const __bf16*>(q_u);
  const __bf16* Kh = reinterpret_cast<const __bf16*>(k_u);
  const __bf16* Vt = reinterpret_cast<const __bf16*>(vt_u);
  __bf16* AO = reinterpret_cast<__bf16*>(ao_u);

  // [0,4096): kbuf (2 x 2048 elems); [4096,8192): vbuf; [8192,10240): ptile x4
  __shared__ __align__(16) __bf16 smem[2 * 2048 + 2 * 2048 + 4 * 512];
  const unsigned int smem_base = (unsigned int)(uintptr_t)(void*)smem;

  const int tid  = threadIdx.x;
  const int lane = tid & 31;
  const int wv   = tid >> 5;
  const int gw   = blockIdx.x * 4 + wv;   // 0..4095
  const int bh   = gw >> 7;               // (b*H + h); block never straddles bh
  const int qt   = gw & 127;
  const int b    = bh >> 4;
  const int h    = bh & 15;
  const int qbase = qt * 16;
  const int col = lane & 15, hi = lane >> 4;

  // Q fragments for this 16-row tile (A operand, K dim = head dim 64)
  const __bf16* qrow = Qh + ((size_t)bh * SS + qbase + col) * DKK;
  const v16bf aq0 = frag_from_bf16(qrow, 0, hi);
  const v16bf aq1 = frag_from_bf16(qrow, 32, hi);

  const __bf16* Kbh  = Kh + (size_t)bh * SS * DKK;   // [S][DK], chunk contiguous
  const __bf16* Vtbh = Vt + (size_t)bh * DKK * SS;   // [DK][S]

  // ---- async-copy issue: waves 0-1 bring K, waves 2-3 bring V --------------
  auto issue_chunk = [&](int buf, int kv0) {
    if (tid < 64) {
      // K chunk: contiguous 4KB; thread t copies 64B at elem offset t*32
      const __bf16* src = Kbh + (size_t)kv0 * DKK + tid * 32;
      unsigned int dst = smem_base + (unsigned int)(buf * 2048 + tid * 32) * 2;
#pragma unroll
      for (int j = 0; j < 4; ++j)
        async_ld_b128(dst + j * 16, src + j * 8);
    } else {
      // V chunk: row dk -> 64B from V^T
      const int dk = tid - 64;
      const __bf16* src = Vtbh + (size_t)dk * SS + kv0;
      unsigned int dst = smem_base + (unsigned int)(4096 + buf * 2048 + dk * 32) * 2;
#pragma unroll
      for (int j = 0; j < 4; ++j)
        async_ld_b128(dst + j * 16, src + j * 8);
    }
  };

  float m_r[8], l_r[8];
  v8f acc[4];
#pragma unroll
  for (int r = 0; r < 8; ++r) { m_r[r] = -3.0e38f; l_r[r] = 0.f; }
#pragma unroll
  for (int t = 0; t < 4; ++t) acc[t] = (v8f){};

  __bf16* ptile = smem + 8192 + wv * 512;

  issue_chunk(0, 0);                       // prologue: chunk 0 -> buf 0

  for (int i = 0; i < SS / 32; ++i) {
    const int cur = i & 1;
    const int kv0 = i * 32;
    const bool more = (i + 1) < (SS / 32);

    if (more) issue_chunk(cur ^ 1, kv0 + 32);   // overlap next chunk's DMA
    wait_async_le(more ? 4 : 0);                // in-order: drains current chunk
    __syncthreads();                            // all waves' copies visible

    const __bf16* kbuf = smem + cur * 2048;            // [32 kv][64 dk]
    const __bf16* vbuf = smem + 4096 + cur * 2048;     // [64 dk][32 kv]

    // ---- scores: two 16x16 tiles, contraction over DK=64 (from LDS) --------
    v8f c0 = {}, c1 = {};
    c0 = wmma_bf16(aq0, frag_from_bf16(kbuf + col * 64, 0,  hi), c0);
    c0 = wmma_bf16(aq1, frag_from_bf16(kbuf + col * 64, 32, hi), c0);
    c1 = wmma_bf16(aq0, frag_from_bf16(kbuf + (col + 16) * 64, 0,  hi), c1);
    c1 = wmma_bf16(aq1, frag_from_bf16(kbuf + (col + 16) * 64, 32, hi), c1);

    // ---- mask + scale (reference: scale first, masked -> -1e9) -------------
    const int kvA = kv0 + col, kvB = kv0 + 16 + col;
    const unsigned char pA = pad[b * SS + kvA];
    const unsigned char pB = pad[b * SS + kvB];
    float s0[8], s1[8];
#pragma unroll
    for (int r = 0; r < 8; ++r) {
      const int qi = qbase + r + 8 * hi;
      const bool a0 = pA || am[(size_t)qi * SS + kvA];
      const bool a1 = pB || am[(size_t)qi * SS + kvB];
      s0[r] = a0 ? c0[r] * SCL : -1e9f;
      s1[r] = a1 ? c1[r] * SCL : -1e9f;
    }

    // ---- online softmax ----------------------------------------------------
    float alpha[8], p0[8], p1[8];
#pragma unroll
    for (int r = 0; r < 8; ++r) {
      float t = half16_max(fmaxf(s0[r], s1[r]));
      const float mn = fmaxf(m_r[r], t);
      alpha[r] = __expf(m_r[r] - mn);
      p0[r] = __expf(s0[r] - mn);
      p1[r] = __expf(s1[r] - mn);
      const float rs = half16_sum(p0[r] + p1[r]);
      l_r[r] = l_r[r] * alpha[r] + rs;
      m_r[r] = mn;
    }
#pragma unroll
    for (int t = 0; t < 4; ++t)
#pragma unroll
      for (int r = 0; r < 8; ++r) acc[t][r] *= alpha[r];

    // ---- P: C-layout -> LDS (16x32 row-major) -> A-layout fragment ---------
#pragma unroll
    for (int r = 0; r < 8; ++r) {
      const int m = r + 8 * hi;
      ptile[m * 32 + col]      = (__bf16)p0[r];
      ptile[m * 32 + 16 + col] = (__bf16)p1[r];
    }
    asm volatile("s_wait_dscnt 0" ::: "memory");
    const v16bf pf = frag_from_bf16(ptile + (lane & 15) * 32, 0, hi);

    // ---- P * V over the 4 DK column tiles (V from LDS) ---------------------
    acc[0] = wmma_bf16(pf, frag_from_bf16(vbuf + (0 * 16 + col) * 32, 0, hi), acc[0]);
    acc[1] = wmma_bf16(pf, frag_from_bf16(vbuf + (1 * 16 + col) * 32, 0, hi), acc[1]);
    acc[2] = wmma_bf16(pf, frag_from_bf16(vbuf + (2 * 16 + col) * 32, 0, hi), acc[2]);
    acc[3] = wmma_bf16(pf, frag_from_bf16(vbuf + (3 * 16 + col) * 32, 0, hi), acc[3]);

    __syncthreads();   // K/V buffer may be overwritten next iteration
  }

  // ---- epilogue: normalize and store merged-head bf16 [B,S,D] --------------
#pragma unroll
  for (int r = 0; r < 8; ++r) {
    const float inv = (l_r[r] > 0.f) ? 1.f / l_r[r] : 0.f;
    const int token = qbase + r + 8 * hi;
    const size_t base = ((size_t)b * SS + token) * DD + h * DKK;
#pragma unroll
    for (int t = 0; t < 4; ++t)
      AO[base + t * 16 + col] = (__bf16)(acc[t][r] * inv);
  }
}

// ===========================================================================
// Kernel 3: out = AO @ wo^T + bo   (bf16 A, fp32 W/out).
// ===========================================================================
__global__ __launch_bounds__(128) void out_proj(
    const unsigned short* __restrict__ ao_u, const float* __restrict__ w,
    const float* __restrict__ bias, float* __restrict__ out) {
  const __bf16* ao = reinterpret_cast<const __bf16*>(ao_u);
  const int lane = threadIdx.x & 31;
  const int wv   = threadIdx.x >> 5;
  const int tile = blockIdx.x * 4 + wv;
  const int mT   = tile >> 5;
  const int nP   = tile & 31;
  const int col  = lane & 15, hi = lane >> 4;

  const __bf16* arow = ao + (size_t)(mT * 16 + col) * DD;
  const float* wrow0 = w + (size_t)(nP * 32 + col) * DD;
  const float* wrow1 = wrow0 + (size_t)16 * DD;

  v8f c0 = {}, c1 = {};
  for (int kc = 0; kc < DD; kc += 32) {
    v16bf a  = frag_from_bf16(arow, kc, hi);
    v16bf b0 = frag_from_f32(wrow0, kc, hi);
    v16bf b1 = frag_from_f32(wrow1, kc, hi);
    c0 = wmma_bf16(a, b0, c0);
    c1 = wmma_bf16(a, b1, c1);
  }
  const float bv0 = bias[nP * 32 + col];
  const float bv1 = bias[nP * 32 + 16 + col];
  const int f0 = nP * 32 + col, f1 = f0 + 16;
#pragma unroll
  for (int r = 0; r < 8; ++r) {
    const int token = mT * 16 + r + 8 * hi;
    out[(size_t)token * DD + f0] = c0[r] + bv0;
    out[(size_t)token * DD + f1] = c1[r] + bv1;
  }
}

// ===========================================================================
// Launch
// ===========================================================================
extern "C" void kernel_launch(void* const* d_in, const int* in_sizes, int n_in,
                              void* d_out, int out_size, void* d_ws, size_t ws_size,
                              hipStream_t stream) {
  const float* q  = (const float*)d_in[0];
  const float* k  = (const float*)d_in[1];
  const float* v  = (const float*)d_in[2];
  const float* wq = (const float*)d_in[3];
  const float* bq = (const float*)d_in[4];
  const float* wk = (const float*)d_in[5];
  const float* bk = (const float*)d_in[6];
  const float* wv = (const float*)d_in[7];
  const float* bv = (const float*)d_in[8];
  const float* wo = (const float*)d_in[9];
  const float* bo = (const float*)d_in[10];
  const unsigned char* pad = (const unsigned char*)d_in[11];
  const unsigned char* am  = (const unsigned char*)d_in[12];

  const size_t nElem = (size_t)BB * HH * SS * DKK;  // = B*S*D
  unsigned short* Qh = (unsigned short*)d_ws;
  unsigned short* Kh = Qh + nElem;
  unsigned short* Vt = Kh + nElem;
  unsigned short* AO = Vt + nElem;

  // 8192 tiles / 4 waves per block
  proj_qkv<<<2048, 128, 0, stream>>>(q, wq, bq, Qh, 0);
  proj_qkv<<<2048, 128, 0, stream>>>(k, wk, bk, Kh, 0);
  proj_qkv<<<2048, 128, 0, stream>>>(v, wv, bv, Vt, 1);

  // 4096 (b,h,qtile) waves / 4 per block
  attn_fwd<<<1024, 128, 0, stream>>>(Qh, Kh, Vt, pad, am, AO);

  out_proj<<<2048, 128, 0, stream>>>(AO, wo, bo, (float*)d_out);
}
